// CODEXReconstruction_32143535243761
// MI455X (gfx1250) — compile-verified, matrix operand order checked by reading, not run on hardware
//
#include <hip/hip_runtime.h>

// ---------------------------------------------------------------------------
// CDNA5 (gfx1250): bf16 WMMA GEMM pipeline, f32 accumulate, TDM tile staging.
// ---------------------------------------------------------------------------

typedef __bf16 bf16;
typedef __attribute__((ext_vector_type(16))) bf16 bf16x16;
typedef __attribute__((ext_vector_type(8)))  bf16 bf16x8;
typedef __attribute__((ext_vector_type(8)))  float f32x8;
typedef __attribute__((ext_vector_type(8)))  unsigned short u16x8;
typedef __attribute__((ext_vector_type(4)))  unsigned int u32x4;
typedef __attribute__((ext_vector_type(8)))  int i32x8;
typedef __attribute__((ext_vector_type(4)))  int i32x4;

#if __has_builtin(__builtin_amdgcn_tensor_load_to_lds) && \
    __has_builtin(__builtin_amdgcn_s_wait_tensorcnt)
#define USE_TDM 1
#else
#define USE_TDM 0
#endif

__device__ __forceinline__ bf16 f2bf(float f) {
  unsigned u = __float_as_uint(f);
  unsigned r = (u + 0x7FFFu + ((u >> 16) & 1u)) >> 16;
  unsigned short h = (unsigned short)r;
  return __builtin_bit_cast(bf16, h);
}
__device__ __forceinline__ float bf2f(unsigned short h) {
  return __uint_as_float(((unsigned)h) << 16);
}

#define BM 128
#define BN 128
#define BK 32
#define LDSTR 40   // padded LDS row stride in elements (80B; TDM pad reproduces it)

#if USE_TDM
// Issue one 2D TDM tile load: tile (tile0 x tile1) elements of 2B data from a
// row-major tensor with dim0-stride `stride0` into LDS at byte addr lds_addr,
// inserting 16B of padding after each 64B row (pad_interval=16dw, amount=4dw).
__device__ __forceinline__ void tdm_load_2d(unsigned lds_addr, const void* gptr,
                                            unsigned tdim0, unsigned tdim1,
                                            unsigned tile0, unsigned tile1,
                                            unsigned long long stride0) {
  const unsigned long long ga = (unsigned long long)(uintptr_t)gptr;
  u32x4 g0;
  g0[0] = 1u;                                              // count=1, user mode
  g0[1] = lds_addr;                                        // lds_addr[31:0]
  g0[2] = (unsigned)(ga & 0xFFFFFFFFu);                    // global_addr[31:0]
  g0[3] = (unsigned)((ga >> 32) & 0x01FFFFFFu) | (2u << 30); // addr[56:32]|type=2

  unsigned long long q0, q1, q2, q3;                       // group1 = 256 bits
  q0 = (1ull << 16)                                        // data_size = 2B
     | (1ull << 20) | (3ull << 22) | (3ull << 25)          // pad: 16dw -> +4dw
     | ((unsigned long long)(tdim0 & 0xFFFFu) << 48);      // tensor_dim0[15:0]
  q1 = (unsigned long long)(tdim0 >> 16)                   // tensor_dim0[31:16]
     | ((unsigned long long)tdim1 << 16)                   // tensor_dim1[31:0]
     | ((unsigned long long)(tile0 & 0xFFFFu) << 48);      // tile_dim0
  q2 = (unsigned long long)(tile1 & 0xFFFFu)               // tile_dim1 (tile2=0)
     | ((stride0 & 0xFFFFFFFFull) << 32);                  // dim0_stride[31:0]
  q3 = (stride0 >> 32) & 0xFFFFull;                        // dim0_stride[47:32]

  i32x8 g1;
  g1[0] = (int)(unsigned)q0; g1[1] = (int)(unsigned)(q0 >> 32);
  g1[2] = (int)(unsigned)q1; g1[3] = (int)(unsigned)(q1 >> 32);
  g1[4] = (int)(unsigned)q2; g1[5] = (int)(unsigned)(q2 >> 32);
  g1[6] = (int)(unsigned)q3; g1[7] = (int)(unsigned)(q3 >> 32);
  i32x4 z4 = {0, 0, 0, 0};
#if defined(__clang_major__) && (__clang_major__ >= 23)
  i32x8 z8 = {0, 0, 0, 0, 0, 0, 0, 0};
  __builtin_amdgcn_tensor_load_to_lds(g0, g1, z4, z4, z8, 0);
#else
  __builtin_amdgcn_tensor_load_to_lds(g0, g1, z4, z4, 0);
#endif
}
#endif  // USE_TDM

// AF32: A operand is f32 (convert while staging); else A is bf16 (TDM staging).
// EPI==0: C(bf16) = relu(A@B + bias)
// EPI==2: final: C(f32); cols < N/2 raw, cols >= N/2 softplus(x)+1e-3
template <bool AF32, int EPI>
__global__ __launch_bounds__(256)
void gemm_bf16_wmma(const void* __restrict__ Ax, const void* __restrict__ Bx,
                    const float* __restrict__ bias, void* __restrict__ Cx,
                    int M, int N, int K) {
  __shared__ __align__(16) bf16 lds_a[BM * LDSTR];
  __shared__ __align__(16) bf16 lds_b[BN * LDSTR];

  const int tid  = threadIdx.x;
  const int lane = tid & 31;
  const int wave = tid >> 5;
  const int wm   = wave & 3;    // M strip of 32 rows
  const int wn   = wave >> 2;   // N strip of 64 cols
  const int l15  = lane & 15;
  const int lhi  = lane >> 4;
  const int koff = lhi << 3;

  const int mbase = blockIdx.y * BM;
  const int nbase = blockIdx.x * BN;

  f32x8 acc[2][4] = {};

  const int arow  = tid >> 1;
  const int ahalf = (tid & 1) << 4;
  const int bn    = tid & 127;
  const int bkh   = (tid >> 7) << 4;
  const int bng   = nbase + bn;

  const float*          Af = (const float*)Ax;
  const unsigned short* Ab = (const unsigned short*)Ax;
  const unsigned short* Bu = (const unsigned short*)Bx;

  for (int k0 = 0; k0 < K; k0 += BK) {
    if (k0 + BK < K) {
      if (AF32)
        __builtin_prefetch(Af + (size_t)(mbase + arow) * K + (k0 + BK + ahalf), 0, 3);
      __builtin_prefetch(Bu + (size_t)(k0 + BK + bkh) * N + (bng < N ? bng : 0), 0, 3);
    }

    // ---- stage A tile into lds_a[m][k] (padded rows) ----
    if constexpr (AF32) {
      float vals[16];
      const size_t abase = (size_t)(mbase + arow) * K + k0 + ahalf;
      if (k0 + ahalf + 16 <= K) {
        const float4 f0 = *(const float4*)(Af + abase);
        const float4 f1 = *(const float4*)(Af + abase + 4);
        const float4 f2 = *(const float4*)(Af + abase + 8);
        const float4 f3 = *(const float4*)(Af + abase + 12);
        vals[0]=f0.x; vals[1]=f0.y; vals[2]=f0.z; vals[3]=f0.w;
        vals[4]=f1.x; vals[5]=f1.y; vals[6]=f1.z; vals[7]=f1.w;
        vals[8]=f2.x; vals[9]=f2.y; vals[10]=f2.z; vals[11]=f2.w;
        vals[12]=f3.x; vals[13]=f3.y; vals[14]=f3.z; vals[15]=f3.w;
      } else {
        #pragma unroll
        for (int i = 0; i < 16; ++i) {
          const int kk = k0 + ahalf + i;
          vals[i] = (kk < K) ? Af[abase + i] : 0.f;
        }
      }
      __align__(16) bf16 pk[16];
      #pragma unroll
      for (int i = 0; i < 16; ++i) pk[i] = f2bf(vals[i]);
      *(bf16x8*)&lds_a[arow * LDSTR + ahalf]     = *(bf16x8*)&pk[0];
      *(bf16x8*)&lds_a[arow * LDSTR + ahalf + 8] = *(bf16x8*)&pk[8];
    } else {
#if USE_TDM
      if (wave == 0) {
        tdm_load_2d((unsigned)(uintptr_t)(void*)lds_a,
                    Ab + (size_t)mbase * K + k0,
                    (unsigned)(K - k0), (unsigned)(M - mbase),
                    (unsigned)BK, (unsigned)BM, (unsigned long long)K);
      }
#else
      const size_t abase = (size_t)(mbase + arow) * K + k0 + ahalf;
      if (k0 + ahalf + 16 <= K) {
        const u16x8 w0 = *(const u16x8*)(Ab + abase);
        const u16x8 w1 = *(const u16x8*)(Ab + abase + 8);
        *(u16x8*)&lds_a[arow * LDSTR + ahalf]     = w0;
        *(u16x8*)&lds_a[arow * LDSTR + ahalf + 8] = w1;
      } else {
        __align__(16) unsigned short pk[16];
        #pragma unroll
        for (int i = 0; i < 16; ++i) {
          const int kk = k0 + ahalf + i;
          pk[i] = (kk < K) ? Ab[abase + i] : (unsigned short)0;
        }
        *(u16x8*)&lds_a[arow * LDSTR + ahalf]     = *(u16x8*)&pk[0];
        *(u16x8*)&lds_a[arow * LDSTR + ahalf + 8] = *(u16x8*)&pk[8];
      }
#endif
    }

    // ---- stage B tile transposed into lds_b[n][k] (bf16 copy) ----
    {
      __align__(16) unsigned short pk[16];
      #pragma unroll
      for (int i = 0; i < 16; ++i) {
        const int kk = k0 + bkh + i;
        pk[i] = (kk < K && bng < N) ? Bu[(size_t)kk * N + bng] : (unsigned short)0;
      }
      *(u16x8*)&lds_b[bn * LDSTR + bkh]     = *(u16x8*)&pk[0];
      *(u16x8*)&lds_b[bn * LDSTR + bkh + 8] = *(u16x8*)&pk[8];
    }

#if USE_TDM
    if constexpr (!AF32) __builtin_amdgcn_s_wait_tensorcnt(0);
#endif
    __syncthreads();

    // ---- fragment loads (2x ds_read_b128 each) + 8 WMMAs ----
    bf16x16 afr[2], bfr[4];
    #pragma unroll
    for (int mi = 0; mi < 2; ++mi) {
      const bf16* p = &lds_a[(wm * 32 + mi * 16 + l15) * LDSTR];
      union { bf16x16 v; bf16x8 h[2]; } u;
      u.h[0] = *(const bf16x8*)(p + koff);
      u.h[1] = *(const bf16x8*)(p + koff + 16);
      afr[mi] = u.v;
    }
    #pragma unroll
    for (int nj = 0; nj < 4; ++nj) {
      const bf16* p = &lds_b[(wn * 64 + nj * 16 + l15) * LDSTR];
      union { bf16x16 v; bf16x8 h[2]; } u;
      u.h[0] = *(const bf16x8*)(p + koff);
      u.h[1] = *(const bf16x8*)(p + koff + 16);
      bfr[nj] = u.v;
    }
    #pragma unroll
    for (int mi = 0; mi < 2; ++mi) {
      #pragma unroll
      for (int nj = 0; nj < 4; ++nj) {
        acc[mi][nj] = __builtin_amdgcn_wmma_f32_16x16x32_bf16(
            false, afr[mi], false, bfr[nj], (short)0, acc[mi][nj], false, false);
      }
    }
    __syncthreads();
  }

  // ---- epilogue ----
  bf16*  Cb = (bf16*)Cx;
  float* Cf = (float*)Cx;
  #pragma unroll
  for (int mi = 0; mi < 2; ++mi) {
    #pragma unroll
    for (int nj = 0; nj < 4; ++nj) {
      const int col = nbase + wn * 64 + nj * 16 + l15;
      if (col >= N) continue;
      const float bv = bias[col];
      #pragma unroll
      for (int j = 0; j < 8; ++j) {
        const int row = mbase + wm * 32 + mi * 16 + (lhi << 3) + j;
        const float v = acc[mi][nj][j] + bv;
        if (EPI == 0) {
          Cb[(size_t)row * N + col] = f2bf(fmaxf(v, 0.f));
        } else {
          const int dim = N >> 1;
          if (col < dim) {
            Cf[(size_t)row * N + col] = v;
          } else {
            const float sp = (v > 20.f) ? v : log1pf(expf(v));
            Cf[(size_t)row * N + col] = sp + 0.001f;
          }
        }
      }
    }
  }
}

// ---------------------------------------------------------------------------
// f32 -> bf16 bulk weight conversion (sizes are multiples of 4)
// ---------------------------------------------------------------------------
__global__ void convert_f32_bf16(const float* __restrict__ src,
                                 bf16* __restrict__ dst, int n) {
  const int i = (blockIdx.x * blockDim.x + threadIdx.x) * 4;
  if (i + 4 <= n) {
    const float4 f = *(const float4*)(src + i);
    dst[i + 0] = f2bf(f.x);
    dst[i + 1] = f2bf(f.y);
    dst[i + 2] = f2bf(f.z);
    dst[i + 3] = f2bf(f.w);
  }
}

// ---------------------------------------------------------------------------
// Gating counts (deterministic integer atomics)
// ---------------------------------------------------------------------------
__global__ void zero_cnt_kernel(int* cnt) {
  if (threadIdx.x < 32) cnt[threadIdx.x] = 0;
}

__global__ void count_kernel(const int* __restrict__ treat, int* __restrict__ cnt,
                             int batch) {
  const int b = blockIdx.x * blockDim.x + threadIdx.x;
  if (b >= batch) return;
  const int t0 = treat[b * 2 + 0];
  const int t1 = treat[b * 2 + 1];
  if (t0 >= 1 && t0 <= 20) atomicAdd(&cnt[t0 - 1], 1);
  if (t1 != t0 && t1 >= 1 && t1 <= 20) atomicAdd(&cnt[t1 - 1], 1);
}

// ---------------------------------------------------------------------------
// Gated expert layer: per-row, <=2 experts; T_W (5 MB f32) stays L2-resident.
// ---------------------------------------------------------------------------
__global__ __launch_bounds__(256)
void expert_kernel(const unsigned short* __restrict__ emb,   // bf16
                   const int* __restrict__ treat, const int* __restrict__ cnt,
                   const float* __restrict__ T_W, const float* __restrict__ T_b,
                   bf16* __restrict__ latent) {
  __shared__ float semb[256];
  const int b = blockIdx.x;
  const int e = threadIdx.x;
  semb[e] = bf2f(emb[(size_t)b * 256 + e]);
  __syncthreads();

  const int t0 = treat[b * 2 + 0];
  const int t1 = treat[b * 2 + 1];
  float acc = 0.f;
  #pragma unroll 1
  for (int c = 0; c < 2; ++c) {
    const int tv = (c == 0) ? t0 : t1;
    if (c == 1 && tv == t0) continue;
    if (tv < 1 || tv > 20) continue;
    if (cnt[tv - 1] <= 1) continue;
    const float* w = T_W + (size_t)(tv - 1) * 256 * 256 + e;
    float p = T_b[(tv - 1) * 256 + e];
    #pragma unroll 8
    for (int k = 0; k < 256; ++k) p += semb[k] * w[(size_t)k * 256];
    acc += fmaxf(p, 0.f);
  }
  latent[(size_t)b * 256 + e] = f2bf(acc);
}

// ---------------------------------------------------------------------------
// Host-side launch
// ---------------------------------------------------------------------------
extern "C" void kernel_launch(void* const* d_in, const int* in_sizes, int n_in,
                              void* d_out, int out_size, void* d_ws, size_t ws_size,
                              hipStream_t stream) {
  (void)in_sizes; (void)n_in; (void)out_size; (void)ws_size;

  const float* input  = (const float*)d_in[0];
  const int*   treat  = (const int*)d_in[1];
  const float* enc_W1 = (const float*)d_in[2];
  const float* enc_b1 = (const float*)d_in[3];
  const float* enc_W2 = (const float*)d_in[4];
  const float* enc_b2 = (const float*)d_in[5];
  const float* T_W    = (const float*)d_in[6];
  const float* T_b    = (const float*)d_in[7];
  const float* dec_W1 = (const float*)d_in[8];
  const float* dec_b1 = (const float*)d_in[9];
  const float* dec_W2 = (const float*)d_in[10];
  const float* dec_b2 = (const float*)d_in[11];
  const float* dec_W3 = (const float*)d_in[12];
  const float* dec_b3 = (const float*)d_in[13];
  float* out = (float*)d_out;

  const int BATCH = 8192;
  const int IN_F  = 5000;

  // workspace carve-up: bf16 weights + bf16 intermediates (~51 MB, L2-resident)
  char* ws = (char*)d_ws;
  int*  cnt = (int*)ws;                                   // 32 ints
  bf16* wb1 = (bf16*)(ws + 256);                          // 5000x512
  bf16* wb2 = wb1 + (size_t)IN_F * 512;                   // 512x256
  bf16* wb3 = wb2 + (size_t)512 * 256;                    // 256x512
  bf16* wb4 = wb3 + (size_t)256 * 512;                    // 512x512
  bf16* wb5 = wb4 + (size_t)512 * 512;                    // 512x10000
  bf16* h1  = wb5 + (size_t)512 * 2 * IN_F;               // 8192x512
  bf16* emb = h1 + (size_t)BATCH * 512;                   // 8192x256
  bf16* lat = emb + (size_t)BATCH * 256;                  // 8192x256
  bf16* h3  = lat + (size_t)BATCH * 256;                  // 8192x512
  bf16* h4  = h3 + (size_t)BATCH * 512;                   // 8192x512

  // weight conversion (one pass per launch; ~35 MB read)
  auto cvt = [&](const float* s, bf16* d, int n) {
    convert_f32_bf16<<<(n / 4 + 255) / 256, 256, 0, stream>>>(s, d, n);
  };
  cvt(enc_W1, wb1, IN_F * 512);
  cvt(enc_W2, wb2, 512 * 256);
  cvt(dec_W1, wb3, 256 * 512);
  cvt(dec_W2, wb4, 512 * 512);
  cvt(dec_W3, wb5, 512 * 2 * IN_F);

  // gating counts
  zero_cnt_kernel<<<1, 32, 0, stream>>>(cnt);
  count_kernel<<<(BATCH + 255) / 256, 256, 0, stream>>>(treat, cnt, BATCH);

  // encoder
  gemm_bf16_wmma<true, 0><<<dim3(512 / BN, BATCH / BM), 256, 0, stream>>>(
      input, wb1, enc_b1, h1, BATCH, 512, IN_F);
  gemm_bf16_wmma<false, 0><<<dim3(256 / BN, BATCH / BM), 256, 0, stream>>>(
      h1, wb2, enc_b2, emb, BATCH, 256, 512);

  // gated experts
  expert_kernel<<<BATCH, 256, 0, stream>>>((const unsigned short*)emb, treat, cnt,
                                           T_W, T_b, lat);

  // decoder
  gemm_bf16_wmma<false, 0><<<dim3(512 / BN, BATCH / BM), 256, 0, stream>>>(
      lat, wb3, dec_b1, h3, BATCH, 512, 256);
  gemm_bf16_wmma<false, 0><<<dim3(512 / BN, BATCH / BM), 256, 0, stream>>>(
      h3, wb4, dec_b2, h4, BATCH, 512, 512);
  gemm_bf16_wmma<false, 2><<<dim3((2 * IN_F + BN - 1) / BN, BATCH / BM), 256, 0, stream>>>(
      h4, wb5, dec_b3, out, BATCH, 2 * IN_F, 512);
}
